// QuantizedLinearWithLoRA_3006477107229
// MI455X (gfx1250) — compile-verified, matrix operand order checked by reading, not run on hardware
//
#include <hip/hip_runtime.h>

typedef __attribute__((ext_vector_type(16))) _Float16 v16h;
typedef __attribute__((ext_vector_type(8)))  _Float16 v8h;
typedef __attribute__((ext_vector_type(4)))  _Float16 v4h;
typedef __attribute__((ext_vector_type(2)))  _Float16 v2h;
typedef __attribute__((ext_vector_type(8)))  float    v8f;
typedef __attribute__((ext_vector_type(4)))  int      v4i;

#define ZERO_POINT   128
#define LORA_SCALING 0.25f     /* alpha/r = 16/64 */
#define D_IN   8192
#define D_OUT  8192
#define BATCH  64
#define RANK   64

#define KSLICE 512             /* K per block (grid.y = 16)             */
#define KSTEP  32              /* K per WMMA                            */
#define NSTEPS (KSLICE / KSTEP)   /* 16, even                           */
#define BLK_N  128             /* 8 waves x 16 columns                  */
#define LDS_STRIDE 520         /* halves per x row: 512 + 8 pad         */

#define NTL(p) __builtin_nontemporal_load(p)

// ---------------------------------------------------------------------------
// f16 magic-number dequant: as_f16(0x6400 | q) == 1024 + q exactly for
// q in [0,255]; subtract 1152 -> (q - 128), exact.  3 VALU per 2 elements.
__device__ __forceinline__ v2h deq2(int q0, int q1) {
    const unsigned pk = ((unsigned)q1 << 16) | (unsigned)q0 | 0x64006400u;
    v2h h = __builtin_bit_cast(v2h, pk);
    return h - (_Float16)1152.0f;
}

__device__ __forceinline__ v8h deq8(v4i a, v4i b) {
    v4h lo = __builtin_shufflevector(deq2(a.x, a.y), deq2(a.z, a.w), 0,1,2,3);
    v4h hi = __builtin_shufflevector(deq2(b.x, b.y), deq2(b.z, b.w), 0,1,2,3);
    return __builtin_shufflevector(lo, hi, 0,1,2,3,4,5,6,7);
}

__device__ __forceinline__ v16h cat8(v8h lo, v8h hi) {
    return __builtin_shufflevector(lo, hi,
        0,1,2,3,4,5,6,7,8,9,10,11,12,13,14,15);
}

struct WQuad { v4i a, b, c, d; };

__device__ __forceinline__ WQuad load_w(const int* __restrict__ p) {
    WQuad w;
    w.a = NTL((const v4i*)(p + 0));
    w.b = NTL((const v4i*)(p + 4));
    w.c = NTL((const v4i*)(p + 16));
    w.d = NTL((const v4i*)(p + 20));
    return w;
}

__device__ __forceinline__ v16h deqw(const WQuad& w) {
    return cat8(deq8(w.a, w.b), deq8(w.c, w.d));
}

// load 4 A-fragments (M-tiles 0..3) for k-step `ks` into distinct registers
__device__ __forceinline__ void load_afrags(v16h af[4],
                                            const _Float16* __restrict__ ldsx,
                                            int col, int khalf, int ks) {
    #pragma unroll
    for (int t = 0; t < 4; ++t) {
        const _Float16* ab = &ldsx[(t * 16 + col) * LDS_STRIDE + (ks << 5) + khalf];
        v8h lo = *(const v8h*)(ab);           // ds_load_b128
        v8h hi = *(const v8h*)(ab + 16);      // ds_load_b128
        af[t] = cat8(lo, hi);
    }
}

// ---------------------------------------------------------------------------
// k0: zero the LoRA mid scratch (64x64 f32)
__global__ void zero_mid_kernel(float* __restrict__ mid) {
    int idx = blockIdx.x * blockDim.x + threadIdx.x;
    if (idx < BATCH * RANK) mid[idx] = 0.0f;
}

// ---------------------------------------------------------------------------
// k1: mid[b][r] += (alpha/r) * sum_k x[b][k]*A[r][k]  over a K chunk of 128
__global__ void lora_mid_kernel(const float* __restrict__ x,
                                const float* __restrict__ A,
                                float* __restrict__ mid) {
    const int kc  = blockIdx.x * 128;
    const int tid = threadIdx.x;
    for (int p = tid; p < BATCH * RANK; p += 256) {
        const int b = p >> 6;
        const int r = p & (RANK - 1);
        const float* xp = x + (size_t)b * D_IN + kc;
        const float* ap = A + (size_t)r * D_IN + kc;
        float acc = 0.0f;
        #pragma unroll
        for (int k = 0; k < 128; k += 4) {
            float4 xv = *(const float4*)(xp + k);
            float4 av = *(const float4*)(ap + k);
            acc += xv.x * av.x + xv.y * av.y + xv.z * av.z + xv.w * av.w;
        }
        atomicAdd(mid + p, LORA_SCALING * acc);
    }
}

// ---------------------------------------------------------------------------
// k2: out[b][o] = bias[o] + sum_r mid[b][r] * lora_B[o][r]
//     (writes out exactly once -> base value for the GEMM atomics)
__global__ void epilogue_init_kernel(const float* __restrict__ mid,
                                     const float* __restrict__ loraB,
                                     const float* __restrict__ bias,
                                     float* __restrict__ out) {
    const int idx = blockIdx.x * blockDim.x + threadIdx.x;   // b*8192 + o
    const int b = idx >> 13;
    const int o = idx & (D_OUT - 1);
    const float* Brow = loraB + (size_t)o * RANK;
    const float* Mrow = mid + b * RANK;
    float acc = bias[o];
    #pragma unroll
    for (int r = 0; r < RANK; r += 4) {
        float4 bb = *(const float4*)(Brow + r);
        float4 mm = *(const float4*)(Mrow + r);
        acc += bb.x * mm.x + bb.y * mm.y + bb.z * mm.z + bb.w * mm.w;
    }
    out[idx] = acc;
}

// ---------------------------------------------------------------------------
// k3: main dequant-GEMM.  grid = (D_OUT/128, D_IN/512), block = 256 (8 waves).
// Wave w owns columns [nbase + 16w, +16); full M=64 as 4 WMMA row-tiles.
// out[b][o] += scale[o] * sum_k x_f16[b][k] * (Wq[o][k]-128)_f16
__global__ void __launch_bounds__(256)
qlora_gemm_kernel(const float* __restrict__ x,
                  const int*   __restrict__ Wq,
                  const float* __restrict__ scale,
                  float* __restrict__ out) {
    extern __shared__ _Float16 ldsx[];            // [64][LDS_STRIDE] f16

    const int tid   = threadIdx.x;
    const int lane  = tid & 31;
    const int wave  = tid >> 5;
    const int kbase = blockIdx.y * KSLICE;
    const int nbase = blockIdx.x * BLK_N;

    // ---- stage x[0:64, kbase:kbase+512) into LDS as f16 (one-time convert)
    #pragma unroll
    for (int i = 0; i < 32; ++i) {
        const int idx = tid + (i << 8);           // 0..8191 float4 units
        const int row = idx >> 7;                 // 128 float4 per row
        const int c4  = (idx & 127) << 2;
        const float4 f = *(const float4*)(x + (size_t)row * D_IN + kbase + c4);
        v4h h; h[0] = (_Float16)f.x; h[1] = (_Float16)f.y;
               h[2] = (_Float16)f.z; h[3] = (_Float16)f.w;
        *(v4h*)(&ldsx[row * LDS_STRIDE + c4]) = h;   // ds_store_b64
    }
    __syncthreads();

    const int col   = lane & 15;                  // column within 16-wide tile
    const int o     = nbase + wave * 16 + col;    // output channel
    const int khalf = (lane >> 4) << 3;           // 0 or 8 (wmma half-lane split)
    const int* wrow = Wq + (size_t)o * D_IN + kbase + khalf;

    v8f acc[4] = {};                              // 4 M-tiles x (16x16 f32)

    // ping-pong pipeline: no register copies between stages
    WQuad w0 = load_w(wrow);
    v16h afA[4], afB[4];
    load_afrags(afA, ldsx, col, khalf, 0);

    for (int ks = 0; ks < NSTEPS; ks += 2) {
        // long-range prefetch of the W stream into GL2
        __builtin_prefetch(wrow + (ks + 4) * KSTEP, 0, 0);

        // even stage: load ks+1 into B-slot, compute ks from A-slot
        WQuad w1 = load_w(wrow + (ks + 1) * KSTEP);
        load_afrags(afB, ldsx, col, khalf, ks + 1);
        v16h bfrag = deqw(w0);
        #pragma unroll
        for (int t = 0; t < 4; ++t)
            acc[t] = __builtin_amdgcn_wmma_f32_16x16x32_f16(
                false, afA[t], false, bfrag, (short)0, acc[t], false, false);

        // odd stage: load ks+2 (clamped) into A-slot, compute ks+1 from B-slot
        const int ks2 = (ks + 2 < NSTEPS) ? ks + 2 : ks + 1;
        w0 = load_w(wrow + ks2 * KSTEP);
        load_afrags(afA, ldsx, col, khalf, ks2);
        bfrag = deqw(w1);
        #pragma unroll
        for (int t = 0; t < 4; ++t)
            acc[t] = __builtin_amdgcn_wmma_f32_16x16x32_f16(
                false, afB[t], false, bfrag, (short)0, acc[t], false, false);
    }

    // ---- epilogue: per-channel scale, atomic accumulate into out
    const float s    = scale[o];
    const int   moff = (lane >> 4) << 3;          // 0 or 8
    float* outp = out + o;
    #pragma unroll
    for (int t = 0; t < 4; ++t) {
        #pragma unroll
        for (int j = 0; j < 8; ++j) {
            const int m = t * 16 + moff + j;
            atomicAdd(outp + (size_t)m * D_OUT, s * acc[t][j]);
        }
    }
}

// ---------------------------------------------------------------------------
extern "C" void kernel_launch(void* const* d_in, const int* in_sizes, int n_in,
                              void* d_out, int out_size, void* d_ws, size_t ws_size,
                              hipStream_t stream) {
    const float* x      = (const float*)d_in[0];
    const int*   Wq     = (const int*)  d_in[1];
    const float* scale  = (const float*)d_in[2];
    const float* lora_A = (const float*)d_in[3];
    const float* lora_B = (const float*)d_in[4];
    const float* bias   = (const float*)d_in[5];
    float* out = (float*)d_out;
    float* mid = (float*)d_ws;                       // 64*64 f32 = 16 KB scratch

    // k0: zero LoRA mid accumulator
    zero_mid_kernel<<<dim3((BATCH * RANK + 255) / 256), 256, 0, stream>>>(mid);

    // k1: mid = (alpha/r) * x @ A^T   (K split across 64 blocks of 128)
    lora_mid_kernel<<<dim3(D_IN / 128), 256, 0, stream>>>(x, lora_A, mid);

    // k2: out = bias + mid @ B^T      (single full write of out)
    epilogue_init_kernel<<<dim3((BATCH * D_OUT) / 256), 256, 0, stream>>>(
        mid, lora_B, bias, out);

    // k3: out += scale[o] * (x @ (Wq-128)^T)  via f16 WMMA
    const size_t shmem = (size_t)BATCH * LDS_STRIDE * sizeof(_Float16); // 66,560 B
    qlora_gemm_kernel<<<dim3(D_OUT / BLK_N, D_IN / KSLICE), 256, shmem, stream>>>(
        x, Wq, scale, out);
}
